// CirC3D_36051955482666
// MI455X (gfx1250) — matrix operand order repaired and checked by reading, not compiled, not run on hardware
//
#include <hip/hip_runtime.h>

typedef __bf16 bf16;
typedef __bf16 v16bf __attribute__((ext_vector_type(16)));
typedef float  v8f   __attribute__((ext_vector_type(8)));
typedef int    v4i   __attribute__((ext_vector_type(4)));

union ABu { struct { v4i lo; v4i hi; } i; v16bf v; };

// ---------------------------------------------------------------------------
// Repack input x: NCDHW f32 -> NDHWC bf16 with channels zero-padded to CP=32.
// ---------------------------------------------------------------------------
__global__ void x_to_cl_k(const float* __restrict__ x, bf16* __restrict__ out,
                          int N, int C, int cpshift, int D, int H, int W) {
  const int CP = 1 << cpshift;
  const long HW = (long)H * W;
  long total = (long)N * D * HW * CP;
  long i = (long)blockIdx.x * blockDim.x + threadIdx.x;
  long stride = (long)gridDim.x * blockDim.x;
  for (; i < total; i += stride) {
    int ci = (int)(i & (CP - 1));
    long s = i >> cpshift;                 // flattened (n,d,h,w)
    int w = (int)(s % W); long t = s / W;
    int h = (int)(t % H); t /= H;
    int d = (int)(t % D); int n = (int)(t / D);
    float v = (ci < C) ? x[(((long)n * C + ci) * D + d) * HW + (long)h * W + w] : 0.0f;
    out[i] = (bf16)v;
  }
}

// ---------------------------------------------------------------------------
// Expand circulant (or dense) weights into WMMA B-fragment-packed bf16.
// K ordering is tap-major: k = tap*CinP + ci  (CinP = Cin padded to >=32).
// Packed layout: [ktile][ntile][lane][elem]; B mapping: col n = lane&15,
// K = (e>>3)*16 + (lane>>4)*8 + (e&7)  ->  one contiguous 32B load per lane per fragment.
// Circulant: W[p*4+i, q*4+j, tap] = c[p, q, (i-j)&3, tap].
// ---------------------------------------------------------------------------
__global__ void pack_weights_k(const float* __restrict__ w, bf16* __restrict__ wp,
                               int Cin, int cishift, int Cout, int circulant) {
  const int CinP = 1 << cishift;
  const int Ktiles = 27 * (CinP >> 5);
  const int Ntiles = Cout >> 4;
  const long total = (long)Ktiles * Ntiles * 512;
  long idx = (long)blockIdx.x * blockDim.x + threadIdx.x;
  long stride = (long)gridDim.x * blockDim.x;
  for (; idx < total; idx += stride) {
    int e     = (int)(idx & 15);
    int lane  = (int)((idx >> 4) & 31);
    long frag = idx >> 9;
    int ntile = (int)(frag % Ntiles);
    int ktile = (int)(frag / Ntiles);
    int kk  = (ktile << 5) + ((e >> 3) << 4) + ((lane >> 4) << 3) + (e & 7);
    int co  = (ntile << 4) + (lane & 15);
    int tap = kk >> cishift;
    int ci  = kk & (CinP - 1);
    float val = 0.0f;
    if (ci < Cin) {
      if (circulant) {
        int Q = Cin >> 2;
        int p = co >> 2, ii = co & 3, q = ci >> 2, jj = ci & 3;
        int s = (ii - jj) & 3;
        val = w[(((long)(p * Q + q)) * 4 + s) * 27 + tap];
      } else {
        val = w[((long)co * Cin + ci) * 27 + tap];
      }
    }
    wp[idx] = (bf16)val;
  }
}

// ---------------------------------------------------------------------------
// Implicit-GEMM 3x3x3 conv (pad=1, stride=1), NDHWC bf16 in -> NDHWC f32 out.
// One wave computes a 16(M) x 64(Cout) tile; 8 waves/block along M.
// Software-pipelined with a branch-free steady-state loop (tail peeled) and
// pure 32-bit element-offset addressing (all tensors < 2^31 elements); the
// B offset is a running accumulator so no per-iteration 64-bit multiplies.
// ---------------------------------------------------------------------------
__global__ void conv3d_wmma_k(const bf16* __restrict__ X, const bf16* __restrict__ Wp,
                              const float* __restrict__ bias, float* __restrict__ Y,
                              int N, int CinP, int D, int H, int W, int Cout) {
  const int lane  = threadIdx.x & 31;
  const int mtile = blockIdx.x * 8 + (threadIdx.x >> 5);
  const int nq    = blockIdx.y;            // 64 output channels per grid-y
  const int Ntiles = Cout >> 4;
  const int ntb    = nq << 2;
  const int HW  = H * W;
  const int DHW = D * HW;
  const int Mtot = N * DHW;
  const int hi  = lane >> 4;
  const int lo  = lane & 15;
  const int hi8 = hi << 3;
  const int cblk = CinP >> 5;
  const int KT   = 27 * cblk;

  // A row decode (needed only for halo/bounds; stores use m directly)
  int am = mtile * 16 + lo;
  const bool avalid = am < Mtot;
  int amc = avalid ? am : 0;
  int an = amc / DHW; int r1 = amc - an * DHW;
  int ad = r1 / HW;   int r2 = r1 - ad * HW;
  int ah = r2 / W;    int aw = r2 - ah * W;

  v8f acc[4];
#pragma unroll
  for (int j = 0; j < 4; ++j) {
    float bv = bias[(nq << 6) + (j << 4) + lo];
#pragma unroll
    for (int r = 0; r < 8; ++r) acc[j][r] = bv;
  }

  // ---- load-position state machine (32-bit offsets, incremental B offset)
  unsigned bofs = (unsigned)(ntb * 32 + lane) * 16u;   // bf16-element offset
  const unsigned bstep = (unsigned)Ntiles * 512u;      // one ktile of fragments

  int cbl = 0, kdl = 0, khl = 0, kwl = 0;
  unsigned apos = 0; int vmL = 0;
  auto tapsetup = [&]() {
    int zd = ad + kdl - 1, zh = ah + khl - 1, zw = aw + kwl - 1;
    bool vt = avalid && ((unsigned)zd < (unsigned)D) &&
              ((unsigned)zh < (unsigned)H) && ((unsigned)zw < (unsigned)W);
    unsigned sp = (unsigned)(((an * D + zd) * H + zh) * W + zw);
    apos = vt ? sp * (unsigned)CinP + (unsigned)hi8 : (unsigned)hi8;  // safe addr
    vmL  = vt ? -1 : 0;
  };
  tapsetup();

  struct FragT { v4i alo, ahi; int vm; v16bf b0, b1, b2, b3; };

  auto loadfrag = [&](FragT& f) {
    const bf16* p = X + (apos + ((unsigned)cbl << 5));
    f.alo = *(const v4i*)(p);              // elements e0..7  (K = hi8+0..7)
    f.ahi = *(const v4i*)(p + 16);         // elements e8..15 (K = hi8+16..23)
    f.vm  = vmL;
    const v16bf* bp = (const v16bf*)(Wp + bofs);
    f.b0 = bp[0]; f.b1 = bp[32]; f.b2 = bp[64]; f.b3 = bp[96];
    bofs += bstep;
    if (++cbl == cblk) {
      cbl = 0;
      if (++kwl == 3) { kwl = 0; if (++khl == 3) { khl = 0; ++kdl; } }
      tapsetup();
    }
  };

  auto compute = [&](const FragT& f) {
    v4i vmask = { f.vm, f.vm, f.vm, f.vm };
    ABu u;
    u.i.lo = f.alo & vmask;                // deferred boundary mask: the wait
    u.i.hi = f.ahi & vmask;                // for A lands one iteration late
    acc[0] = __builtin_amdgcn_wmma_f32_16x16x32_bf16(false, u.v, false, f.b0,
                                                     (short)0, acc[0], false, false);
    acc[1] = __builtin_amdgcn_wmma_f32_16x16x32_bf16(false, u.v, false, f.b1,
                                                     (short)0, acc[1], false, false);
    acc[2] = __builtin_amdgcn_wmma_f32_16x16x32_bf16(false, u.v, false, f.b2,
                                                     (short)0, acc[2], false, false);
    acc[3] = __builtin_amdgcn_wmma_f32_16x16x32_bf16(false, u.v, false, f.b3,
                                                     (short)0, acc[3], false, false);
  };

  // ---- software-pipelined main loop: branch-free body, peeled tail ----
  FragT f0, f1;
  loadfrag(f0);                            // block 0   (KT >= 27 always)
  int kt = 0;
  for (; kt + 2 < KT; kt += 2) {
    loadfrag(f1);                          // block kt+1 issued ahead
    compute(f0);                           // block kt
    loadfrag(f0);                          // block kt+2 issued ahead
    compute(f1);                           // block kt+1
  }
  if (kt + 1 < KT) {                       // two blocks left (KT even)
    loadfrag(f1);
    compute(f0);
    compute(f1);
  } else {                                 // one block left (KT odd)
    compute(f0);
  }

  // ---- store D: NDHWC means index = m*Cout + co (no decode needed)
#pragma unroll
  for (int r = 0; r < 8; ++r) {
    int m = mtile * 16 + r + hi8;
    if (m < Mtot) {
#pragma unroll
      for (int j = 0; j < 4; ++j) {
        Y[(unsigned)m * (unsigned)Cout + (unsigned)((nq << 6) + (j << 4) + lo)] = acc[j][r];
      }
    }
  }
}

// ---------------------------------------------------------------------------
// BatchNorm statistics (channels-last): one block per channel -> scale/shift.
// ---------------------------------------------------------------------------
__global__ void bn_stats_k(const float* __restrict__ x, const float* __restrict__ g,
                           const float* __restrict__ bb, float* __restrict__ scale,
                           float* __restrict__ shift, int C, long S) {
  const int c = blockIdx.x;
  float s = 0.0f, s2 = 0.0f;
  for (long i = threadIdx.x; i < S; i += blockDim.x) {
    float v = x[i * C + c];
    s += v; s2 += v * v;
  }
  __shared__ float sh[256], sh2[256];
  sh[threadIdx.x] = s; sh2[threadIdx.x] = s2;
  __syncthreads();
  for (int o = 128; o > 0; o >>= 1) {
    if ((int)threadIdx.x < o) {
      sh[threadIdx.x]  += sh[threadIdx.x + o];
      sh2[threadIdx.x] += sh2[threadIdx.x + o];
    }
    __syncthreads();
  }
  if (threadIdx.x == 0) {
    float mean = sh[0] / (float)S;
    float var  = sh2[0] / (float)S - mean * mean;
    float inv  = rsqrtf(var + 1e-5f);
    float sc   = g[c] * inv;
    scale[c] = sc;
    shift[c] = bb[c] - mean * sc;
  }
}

// ---------------------------------------------------------------------------
// BN apply + ReLU (channels-last, C is a power of two), f32 -> bf16
// ---------------------------------------------------------------------------
__global__ void bn_apply_relu_k(const float* __restrict__ x, const float* __restrict__ scale,
                                const float* __restrict__ shift, bf16* __restrict__ y,
                                long total, int cmask) {
  long i = (long)blockIdx.x * blockDim.x + threadIdx.x;
  long stride = (long)gridDim.x * blockDim.x;
  for (; i < total; i += stride) {
    int c = (int)i & cmask;
    float v = fmaf(x[i], scale[c], shift[c]);
    y[i] = (bf16)fmaxf(v, 0.0f);
  }
}

// ---------------------------------------------------------------------------
// 3D max pool, channels-last bf16 (coalesced: channel fastest)
// ---------------------------------------------------------------------------
__global__ void maxpool3d_k(const bf16* __restrict__ in, bf16* __restrict__ out,
                            int N, int C, int clog,
                            int D, int H, int W, int OD, int OH, int OW,
                            int wd, int wh, int ww, int sd, int s_h, int sw,
                            int pd, int ph, int pw) {
  long total = ((long)N * OD * OH * OW) << clog;
  long i = (long)blockIdx.x * blockDim.x + threadIdx.x;
  long stride = (long)gridDim.x * blockDim.x;
  for (; i < total; i += stride) {
    int c = (int)i & (C - 1);
    long s = i >> clog;
    int ow = (int)(s % OW); long t = s / OW;
    int oh = (int)(t % OH); t /= OH;
    int od = (int)(t % OD); int n = (int)(t / OD);
    float m = -__builtin_inff();
    for (int dz = 0; dz < wd; ++dz) {
      int z = od * sd - pd + dz;
      if ((unsigned)z >= (unsigned)D) continue;
      for (int dy = 0; dy < wh; ++dy) {
        int yy = oh * s_h - ph + dy;
        if ((unsigned)yy >= (unsigned)H) continue;
        for (int dx = 0; dx < ww; ++dx) {
          int xx = ow * sw - pw + dx;
          if ((unsigned)xx >= (unsigned)W) continue;
          float v = (float)in[((((long)n * D + z) * H + yy) * W + xx) * C + c];
          m = fmaxf(m, v);
        }
      }
    }
    out[i] = (bf16)m;
  }
}

// ---------------------------------------------------------------------------
// Global average over spatial dims (channels-last), then FC head
// ---------------------------------------------------------------------------
__global__ void avg_k(const bf16* __restrict__ in, float* __restrict__ out,
                      int NC, int C, int S) {
  int i = blockIdx.x * blockDim.x + threadIdx.x;
  if (i < NC) {
    int n = i / C, c = i - n * C;
    float s = 0.0f;
    for (int k = 0; k < S; ++k) s += (float)in[((long)n * S + k) * C + c];
    out[i] = s / (float)S;
  }
}

__global__ void fc_k(const float* __restrict__ a, const float* __restrict__ w,
                     const float* __restrict__ b, float* __restrict__ out,
                     int N, int C, int Kd) {
  int i = blockIdx.x * blockDim.x + threadIdx.x;
  if (i < N * C) {
    int n = i / C, c = i - n * C;
    float s = b[c];
    for (int k = 0; k < Kd; ++k) s = fmaf(a[n * Kd + k], w[(long)c * Kd + k], s);
    out[i] = s;
  }
}

// ---------------------------------------------------------------------------
// Host orchestration
// ---------------------------------------------------------------------------
extern "C" void kernel_launch(void* const* d_in, const int* in_sizes, int n_in,
                              void* d_out, int out_size, void* d_ws, size_t ws_size,
                              hipStream_t stream) {
  (void)in_sizes; (void)n_in; (void)out_size; (void)ws_size;

  const float* x   = (const float*)d_in[0];
  const float* w1  = (const float*)d_in[1];
  const float* b1c = (const float*)d_in[2];
  const float* cw[7];  const float* cb[7];
  for (int i = 0; i < 7; ++i) { cw[i] = (const float*)d_in[3 + 2 * i]; cb[i] = (const float*)d_in[4 + 2 * i]; }
  const float* gg[8]; const float* gb[8];
  for (int i = 0; i < 8; ++i) { gg[i] = (const float*)d_in[17 + 2 * i]; gb[i] = (const float*)d_in[18 + 2 * i]; }
  const float* fcw = (const float*)d_in[33];
  const float* fcb = (const float*)d_in[34];
  float* out = (float*)d_out;

  char* ws = (char*)d_ws;
  size_t off = 0;
  auto alloc = [&](size_t bytes) -> void* {
    off = (off + 255) & ~(size_t)255;
    void* p = ws + off;
    off += bytes;
    return p;
  };

  const long MAXE = 8L * 64 * 16 * 112 * 112;  // largest activation element count
  bf16*  bufA  = (bf16*)alloc(MAXE * sizeof(bf16));
  bf16*  bufB  = (bf16*)alloc(MAXE * sizeof(bf16));
  float* conv  = (float*)alloc(MAXE * sizeof(float));
  float* scale = (float*)alloc(512 * sizeof(float));
  float* shift = (float*)alloc(512 * sizeof(float));
  float* avg   = (float*)alloc(8 * 512 * sizeof(float));

  struct L { const float* w; const float* bias; const float* g; const float* b;
             int Cin, CinP, Cout, circ; };
  L Ls[8] = {
    { w1,    b1c,   gg[0], gb[0],   3,  32,  64, 0 },
    { cw[0], cb[0], gg[1], gb[1],  64,  64, 128, 1 },
    { cw[1], cb[1], gg[2], gb[2], 128, 128, 256, 1 },
    { cw[2], cb[2], gg[3], gb[3], 256, 256, 256, 1 },
    { cw[3], cb[3], gg[4], gb[4], 256, 256, 512, 1 },
    { cw[4], cb[4], gg[5], gb[5], 512, 512, 512, 1 },
    { cw[5], cb[5], gg[6], gb[6], 512, 512, 512, 1 },
    { cw[6], cb[6], gg[7], gb[7], 512, 512, 512, 1 },
  };

  // Expand + pack all weights into WMMA B-fragment order (bf16, tap-major K)
  bf16* wp[8];
  for (int i = 0; i < 8; ++i) {
    int cishift = __builtin_ctz((unsigned)Ls[i].CinP);
    int Kt = 27 * (Ls[i].CinP / 32), Nt = Ls[i].Cout / 16;
    long elems = (long)Kt * Nt * 512;
    wp[i] = (bf16*)alloc(elems * sizeof(bf16));
    int blocks = (int)((elems + 255) / 256);
    pack_weights_k<<<blocks, 256, 0, stream>>>(Ls[i].w, wp[i], Ls[i].Cin, cishift,
                                               Ls[i].Cout, Ls[i].circ);
  }

  auto conv_bn = [&](const bf16* in, bf16* outb, int li, int N, int D, int H, int W) {
    const L& l = Ls[li];
    long M = (long)N * D * H * W;
    int Mtiles = (int)((M + 15) / 16);
    dim3 grid((Mtiles + 7) / 8, l.Cout / 64);
    conv3d_wmma_k<<<grid, 256, 0, stream>>>(in, wp[li], l.bias, conv,
                                            N, l.CinP, D, H, W, l.Cout);
    bn_stats_k<<<l.Cout, 256, 0, stream>>>(conv, l.g, l.b, scale, shift, l.Cout, M);
    long total = M * l.Cout;
    int blocks = (int)((total + 255) / 256); if (blocks > 16384) blocks = 16384;
    bn_apply_relu_k<<<blocks, 256, 0, stream>>>(conv, scale, shift, outb, total, l.Cout - 1);
  };

  auto pool = [&](const bf16* in, bf16* outb, int N, int C, int D, int H, int W,
                  int wd, int wh, int ww, int sd, int s_h, int sw,
                  int pd, int ph, int pw) {
    int OD = (D + 2 * pd - wd) / sd + 1;
    int OH = (H + 2 * ph - wh) / s_h + 1;
    int OW = (W + 2 * pw - ww) / sw + 1;
    long total = (long)N * C * OD * OH * OW;
    int blocks = (int)((total + 255) / 256); if (blocks > 16384) blocks = 16384;
    int clog = __builtin_ctz((unsigned)C);
    maxpool3d_k<<<blocks, 256, 0, stream>>>(in, outb, N, C, clog, D, H, W, OD, OH, OW,
                                            wd, wh, ww, sd, s_h, sw, pd, ph, pw);
  };

  // input -> channels-last bf16, padded to 32 channels
  x_to_cl_k<<<8192, 256, 0, stream>>>(x, bufA, 8, 3, 5, 16, 112, 112);

  conv_bn(bufA, bufB, 0, 8, 16, 112, 112);                 // (8,16,112,112,64)
  pool(bufB, bufA, 8, 64, 16, 112, 112, 1, 2, 2, 1, 2, 2, 0, 0, 0);   // (8,16,56,56,64)
  conv_bn(bufA, bufB, 1, 8, 16, 56, 56);                   // (8,16,56,56,128)
  pool(bufB, bufA, 8, 128, 16, 56, 56, 2, 2, 2, 2, 2, 2, 0, 0, 0);    // (8,8,28,28,128)
  conv_bn(bufA, bufB, 2, 8, 8, 28, 28);                    // (8,8,28,28,256)
  conv_bn(bufB, bufA, 3, 8, 8, 28, 28);                    // (8,8,28,28,256)
  pool(bufA, bufB, 8, 256, 8, 28, 28, 2, 2, 2, 2, 2, 2, 0, 0, 0);     // (8,4,14,14,256)
  conv_bn(bufB, bufA, 4, 8, 4, 14, 14);                    // (8,4,14,14,512)
  conv_bn(bufA, bufB, 5, 8, 4, 14, 14);                    // (8,4,14,14,512)
  pool(bufB, bufA, 8, 512, 4, 14, 14, 2, 2, 2, 2, 2, 2, 0, 0, 0);     // (8,2,7,7,512)
  conv_bn(bufA, bufB, 6, 8, 2, 7, 7);                      // (8,2,7,7,512)
  conv_bn(bufB, bufA, 7, 8, 2, 7, 7);                      // (8,2,7,7,512)
  pool(bufA, bufB, 8, 512, 2, 7, 7, 2, 2, 2, 2, 2, 2, 0, 1, 1);       // (8,1,4,4,512)

  avg_k<<<(8 * 512 + 255) / 256, 256, 0, stream>>>(bufB, avg, 8 * 512, 512, 16);
  fc_k<<<(8 * 101 + 255) / 256, 256, 0, stream>>>(avg, fcw, fcb, out, 8, 101, 512);
}
